// InsertionEdgeHead_48352741818843
// MI455X (gfx1250) — matrix operand order repaired and checked by reading, not compiled, not run on hardware
//
#include <hip/hip_runtime.h>

typedef __attribute__((ext_vector_type(16))) _Float16 v16h;
typedef __attribute__((ext_vector_type(8)))  float    v8f;

#define HID       128
#define H2        64
#define K_GMM     8
#define N_ATOM    16
#define N_CHARGE  7
#define NUM_RBF   16
#define IN_DIM    295
#define IN_PAD    320     // 10 chunks of 32
#define N_EDGE_T  5

#define W1_CH 10
#define W1_NT 8
#define W2_CH 4
#define W2_NT 4
#define W3_CH 2
#define W3_NT 1
// each (chunk,ntile) B-fragment block = 32 lanes * 16 halves = 512 halves
#define FRAG_HALVES 512

// ---------------------------------------------------------------------------
// Weight repack: f32 row-major -> f16 WMMA B-fragment layout.
// B (32x16, KxN) per-lane layout (wave32): lane L holds column N = L%16;
// half i (0..15) holds K = (i&7) + (i>=8 ? 16 : 0) + (L>=16 ? 8 : 0) within the 32-deep chunk.
// Stored so each lane's 16 halves are contiguous (2x b128 load per fragment).
// ---------------------------------------------------------------------------
__global__ void pack_weights_kernel(const float* __restrict__ W1,
                                    const float* __restrict__ W2,
                                    const float* __restrict__ W3,
                                    _Float16* __restrict__ w1p,
                                    _Float16* __restrict__ w2p,
                                    _Float16* __restrict__ w3p) {
  int gid = blockIdx.x * 256 + threadIdx.x;
  const int t1 = W1_CH * W1_NT * FRAG_HALVES;   // 40960
  const int t2 = W2_CH * W2_NT * FRAG_HALVES;   // 8192
  const int t3 = W3_CH * W3_NT * FRAG_HALVES;   // 1024
  if (gid < t1) {
    int i = gid & 15, lane = (gid >> 4) & 31, tile = gid >> 9;
    int ch = tile / W1_NT, nt = tile % W1_NT;
    int k = ch * 32 + (i & 7) + ((i >> 3) << 4) + ((lane >= 16) ? 8 : 0);
    int n = nt * 16 + (lane & 15);
    float v = (k < IN_DIM) ? W1[k * HID + n] : 0.0f;
    w1p[gid] = (_Float16)v;
  } else if (gid < t1 + t2) {
    int g = gid - t1;
    int i = g & 15, lane = (g >> 4) & 31, tile = g >> 9;
    int ch = tile / W2_NT, nt = tile % W2_NT;
    int k = ch * 32 + (i & 7) + ((i >> 3) << 4) + ((lane >= 16) ? 8 : 0);
    int n = nt * 16 + (lane & 15);
    w2p[g] = (_Float16)W2[k * H2 + n];
  } else if (gid < t1 + t2 + t3) {
    int g = gid - t1 - t2;
    int i = g & 15, lane = (g >> 4) & 31, ch = g >> 9;
    int k = ch * 32 + (i & 7) + ((i >> 3) << 4) + ((lane >= 16) ? 8 : 0);
    int n = lane & 15;
    float v = (n < N_EDGE_T && k < H2) ? W3[k * N_EDGE_T + n] : 0.0f;
    w3p[g] = (_Float16)v;
  }
}

// ---------------------------------------------------------------------------
// A-fragment (16x32 f16) from row-major f16 LDS tile.
// Lanes 0-15: M=lane, K = {kb+0..7, kb+16..23}; lanes 16-31: kb += 8.
// ---------------------------------------------------------------------------
__device__ __forceinline__ v16h load_afrag(const _Float16* lds, int stride,
                                           int lane, int chunk) {
  int row = lane & 15;
  int kb  = chunk * 32 + ((lane >= 16) ? 8 : 0);
  const uint32_t* q = (const uint32_t*)(lds + row * stride + kb);
  union { v16h v; uint32_t u[8]; } r;
  r.u[0] = q[0]; r.u[1] = q[1]; r.u[2] = q[2]; r.u[3] = q[3];
  q = (const uint32_t*)(lds + row * stride + kb + 16);
  r.u[4] = q[0]; r.u[5] = q[1]; r.u[6] = q[2]; r.u[7] = q[3];
  return r.v;
}

__device__ __forceinline__ uint32_t hash_u32(uint32_t a) {
  a *= 0x9E3779B9u;
  a ^= a >> 16; a *= 0x7feb352du;
  a ^= a >> 15; a *= 0x846ca68bu;
  a ^= a >> 16;
  return a;
}

// ---------------------------------------------------------------------------
// Main fused kernel: 1 block = 16 edges, 128 threads = 4 waves (wave32).
// ---------------------------------------------------------------------------
__global__ __launch_bounds__(128) void insertion_edge_head_kernel(
    const float* __restrict__ h,   const float* __restrict__ x,
    const float* __restrict__ mu,  const float* __restrict__ pi,
    const float* __restrict__ ap,  const float* __restrict__ cp,
    const float* __restrict__ b1,  const float* __restrict__ g1,
    const float* __restrict__ be1, const float* __restrict__ b2,
    const float* __restrict__ g2,  const float* __restrict__ be2,
    const float* __restrict__ b3,
    const int* __restrict__ spawn, const int* __restrict__ target,
    const _Float16* __restrict__ w1p, const _Float16* __restrict__ w2p,
    const _Float16* __restrict__ w3p,
    float* __restrict__ out, int E) {
  __shared__ _Float16 featA[16 * IN_PAD];   // 10 KB
  __shared__ float    zbuf[16 * HID];       // 8 KB (reused by GEMM2)
  __shared__ _Float16 z1h[16 * HID];        // 4 KB
  __shared__ _Float16 z2h[16 * H2];         // 2 KB
  __shared__ int   sS[16], sT[16], sKk[16];
  __shared__ float sD[16], sM[16], sR[16];

  const int tid  = threadIdx.x;
  const int lane = tid & 31;
  const int wave = tid >> 5;
  const int e0   = blockIdx.x * 16;

  // ---- Phase A1: per-edge scalars (GMM component sample + distance) ----
  if (tid < 16) {
    int e  = e0 + tid;
    int ec = (e < E) ? e : 0;
    int s  = spawn[ec], tg = target[ec];
    int kbest = 0; float best = -3.0e38f;
    for (int j = 0; j < K_GMM; ++j) {
      uint32_t r = hash_u32((uint32_t)(ec * K_GMM + j) ^ 0xC0FFEE42u);
      float u   = ((r >> 8) + 0.5f) * (1.0f / 16777216.0f);
      float gum = -__logf(-__logf(u));
      float sc  = __logf(pi[s * K_GMM + j] + 1e-9f) + gum;
      if (sc > best) { best = sc; kbest = j; }
    }
    float dx = mu[(s * K_GMM + kbest) * 3 + 0] - x[tg * 3 + 0];
    float dy = mu[(s * K_GMM + kbest) * 3 + 1] - x[tg * 3 + 1];
    float dz = mu[(s * K_GMM + kbest) * 3 + 2] - x[tg * 3 + 2];
    float d  = fmaxf(sqrtf(dx * dx + dy * dy + dz * dz), 1e-6f);
    sS[tid] = s; sT[tid] = tg; sKk[tid] = kbest; sD[tid] = d;
  }
  __syncthreads();

  // ---- Phase A2: build 16x320 f16 feature tile in LDS ----
  // 5120 halves = 2560 dwords; 128 threads x 20 iters, pairwise columns.
  for (int it = 0; it < 20; ++it) {
    int j    = it * 128 + tid;
    int row  = j / 160;
    int col  = (j % 160) * 2;
    int s = sS[row], tg = sT[row], kk = sKk[row];
    float d = sD[row];
    float v[2];
#pragma unroll
    for (int p = 0; p < 2; ++p) {
      int c = col + p;
      float val;
      if (c < HID)            val = h[s * HID + c];
      else if (c < 2 * HID)   val = h[tg * HID + (c - HID)];
      else if (c < 272) {
        int cc = c - 256;
        float t = (d - (float)cc * (10.0f / 15.0f)) * (1.0f / 0.625f);
        val = __expf(-0.5f * t * t);
      }
      else if (c < 288)       val = ap[(s * K_GMM + kk) * N_ATOM + (c - 272)];
      else if (c < IN_DIM)    val = cp[(s * K_GMM + kk) * N_CHARGE + (c - 288)];
      else                    val = 0.0f;
      v[p] = val;
    }
    featA[row * IN_PAD + col]     = (_Float16)v[0];
    featA[row * IN_PAD + col + 1] = (_Float16)v[1];
  }
  __syncthreads();

  // ---- Phase B: GEMM1 [16,320] x [320,128] ; wave owns 2 N-tiles ----
  {
    v8f acc0 = {}; v8f acc1 = {};
    const int nt0 = wave * 2, nt1 = wave * 2 + 1;
    for (int c = 0; c < W1_CH; ++c) {
      v16h a  = load_afrag(featA, IN_PAD, lane, c);
      v16h bf0 = *(const v16h*)(w1p + ((c * W1_NT + nt0) * 32 + lane) * 16);
      v16h bf1 = *(const v16h*)(w1p + ((c * W1_NT + nt1) * 32 + lane) * 16);
      acc0 = __builtin_amdgcn_wmma_f32_16x16x32_f16(false, a, false, bf0,
                                                    (short)0, acc0, false, false);
      acc1 = __builtin_amdgcn_wmma_f32_16x16x32_f16(false, a, false, bf1,
                                                    (short)0, acc1, false, false);
    }
    int n0 = nt0 * 16 + (lane & 15);
    int n1 = nt1 * 16 + (lane & 15);
    float bb0 = b1[n0], bb1 = b1[n1];
    int mbase = (lane >> 4) * 8;
#pragma unroll
    for (int v = 0; v < 8; ++v) {
      zbuf[(v + mbase) * HID + n0] = acc0[v] + bb0;
      zbuf[(v + mbase) * HID + n1] = acc1[v] + bb1;
    }
  }
  __syncthreads();

  // ---- LN1 + SiLU -> z1h (f16) ----
  if (tid < 16) {
    float m = 0.0f;
    for (int c = 0; c < HID; ++c) m += zbuf[tid * HID + c];
    m *= (1.0f / HID);
    float var = 0.0f;
    for (int c = 0; c < HID; ++c) { float dd = zbuf[tid * HID + c] - m; var += dd * dd; }
    sM[tid] = m; sR[tid] = rsqrtf(var * (1.0f / HID) + 1e-5f);
  }
  __syncthreads();
  for (int it = 0; it < 16; ++it) {
    int j = it * 128 + tid;
    int row = j >> 7, col = j & 127;
    float v = (zbuf[row * HID + col] - sM[row]) * sR[row] * g1[col] + be1[col];
    v = v / (1.0f + __expf(-v));
    z1h[row * HID + col] = (_Float16)v;
  }
  __syncthreads();

  // ---- Phase C: GEMM2 [16,128] x [128,64] ; wave owns 1 N-tile ----
  {
    v8f acc = {};
    for (int c = 0; c < W2_CH; ++c) {
      v16h a = load_afrag(z1h, HID, lane, c);
      v16h b = *(const v16h*)(w2p + ((c * W2_NT + wave) * 32 + lane) * 16);
      acc = __builtin_amdgcn_wmma_f32_16x16x32_f16(false, a, false, b,
                                                   (short)0, acc, false, false);
    }
    int n = wave * 16 + (lane & 15);
    float bb = b2[n];
    int mbase = (lane >> 4) * 8;
#pragma unroll
    for (int v = 0; v < 8; ++v) zbuf[(v + mbase) * H2 + n] = acc[v] + bb;
  }
  __syncthreads();

  // ---- LN2 + SiLU -> z2h (f16) ----
  if (tid < 16) {
    float m = 0.0f;
    for (int c = 0; c < H2; ++c) m += zbuf[tid * H2 + c];
    m *= (1.0f / H2);
    float var = 0.0f;
    for (int c = 0; c < H2; ++c) { float dd = zbuf[tid * H2 + c] - m; var += dd * dd; }
    sM[tid] = m; sR[tid] = rsqrtf(var * (1.0f / H2) + 1e-5f);
  }
  __syncthreads();
  for (int it = 0; it < 8; ++it) {
    int j = it * 128 + tid;
    int row = j >> 6, col = j & 63;
    float v = (zbuf[row * H2 + col] - sM[row]) * sR[row] * g2[col] + be2[col];
    v = v / (1.0f + __expf(-v));
    z2h[row * H2 + col] = (_Float16)v;
  }
  __syncthreads();

  // ---- Phase D: GEMM3 [16,64] x [64,16(pad of 5)] on wave 0 only ----
  if (wave == 0) {
    v8f acc = {};
    for (int c = 0; c < W3_CH; ++c) {
      v16h a = load_afrag(z2h, H2, lane, c);
      v16h b = *(const v16h*)(w3p + (c * 32 + lane) * 16);
      acc = __builtin_amdgcn_wmma_f32_16x16x32_f16(false, a, false, b,
                                                   (short)0, acc, false, false);
    }
    int n = lane & 15;
    if (n < N_EDGE_T) {
      float bb = b3[n];
      int mbase = (lane >> 4) * 8;
#pragma unroll
      for (int v = 0; v < 8; ++v) {
        int e = e0 + v + mbase;
        if (e < E) out[e * N_EDGE_T + n] = acc[v] + bb;
      }
    }
  }
}

extern "C" void kernel_launch(void* const* d_in, const int* in_sizes, int n_in,
                              void* d_out, int out_size, void* d_ws, size_t ws_size,
                              hipStream_t stream) {
  const float* h   = (const float*)d_in[0];
  const float* x   = (const float*)d_in[1];
  const float* mu  = (const float*)d_in[2];
  const float* pi  = (const float*)d_in[3];
  const float* ap  = (const float*)d_in[4];
  const float* cp  = (const float*)d_in[5];
  const float* W1  = (const float*)d_in[6];
  const float* b1  = (const float*)d_in[7];
  const float* g1  = (const float*)d_in[8];
  const float* be1 = (const float*)d_in[9];
  const float* W2  = (const float*)d_in[10];
  const float* b2  = (const float*)d_in[11];
  const float* g2  = (const float*)d_in[12];
  const float* be2 = (const float*)d_in[13];
  const float* W3  = (const float*)d_in[14];
  const float* b3  = (const float*)d_in[15];
  // d_in[16] = batch (unused by reference computation)
  const int* spawn  = (const int*)d_in[17];
  const int* target = (const int*)d_in[18];
  const int E = in_sizes[17];

  _Float16* w1p = (_Float16*)d_ws;
  _Float16* w2p = w1p + W1_CH * W1_NT * FRAG_HALVES;
  _Float16* w3p = w2p + W2_CH * W2_NT * FRAG_HALVES;

  const int packTotal = (W1_CH * W1_NT + W2_CH * W2_NT + W3_CH * W3_NT) * FRAG_HALVES;
  pack_weights_kernel<<<(packTotal + 255) / 256, 256, 0, stream>>>(W1, W2, W3,
                                                                   w1p, w2p, w3p);

  const int blocks = (E + 15) / 16;
  insertion_edge_head_kernel<<<blocks, 128, 0, stream>>>(
      h, x, mu, pi, ap, cp, b1, g1, be1, b2, g2, be2, b3,
      spawn, target, w1p, w2p, w3p, (float*)d_out, E);
}
